// PointNet_50036368998510
// MI455X (gfx1250) — matrix operand order repaired and checked by reading, not compiled, not run on hardware
//
#include <hip/hip_runtime.h>

typedef __attribute__((ext_vector_type(16))) _Float16 v16h;
typedef __attribute__((ext_vector_type(8)))  _Float16 v8h;
typedef __attribute__((ext_vector_type(4)))  _Float16 v4h;
typedef __attribute__((ext_vector_type(8)))  float    v8f;

#define B_GR   64
#define NPG    2048
#define NPTS   (B_GR*NPG)      // 131072
#define KNN    10
#define HDIM   64
#define EPD    11              // edges per dst: 10 knn + self loop
#define DPB    16              // dst nodes per block
#define EDGES  (DPB*EPD)       // 176 = 11 tiles of 16
#define ETILES (EDGES/16)      // 11
#define KDIM   96              // 64 feat + relx + rely + 1(bias) + pad
#define NBLK_CONV (NPTS/DPB)   // 8192
#define EPS_PN 1e-5f

// ---------------------------------------------------------------- kNN ------
__global__ __launch_bounds__(128)
void knn_kernel(const float* __restrict__ pos, int* __restrict__ nbr) {
    __shared__ float sx[128], sy[128];
    const int gb    = blockIdx.x >> 4;     // graph
    const int chunk = blockIdx.x & 15;
    const int tid   = threadIdx.x;
    const int base  = gb * NPG;
    const int qi    = chunk * 128 + tid;   // query index within graph
    const float qx  = pos[2*(base+qi)];
    const float qy  = pos[2*(base+qi)+1];

    float bd[KNN]; int bi[KNN];
#pragma unroll
    for (int k = 0; k < KNN; ++k) { bd[k] = 3.0e38f; bi[k] = 0; }

    for (int t0 = 0; t0 < NPG; t0 += 128) {
        __syncthreads();
        sx[tid] = pos[2*(base+t0+tid)];
        sy[tid] = pos[2*(base+t0+tid)+1];
        __syncthreads();
        for (int j = 0; j < 128; ++j) {
            const int cj = t0 + j;
            const float dx = qx - sx[j], dy = qy - sy[j];
            const float d  = dx*dx + dy*dy;
            if (cj != qi && d < bd[KNN-1]) {
                int p = KNN-1;
                while (p > 0 && bd[p-1] > d) { bd[p]=bd[p-1]; bi[p]=bi[p-1]; --p; }
                bd[p] = d; bi[p] = cj;
            }
        }
    }
#pragma unroll
    for (int k = 0; k < KNN; ++k) nbr[(base+qi)*KNN + k] = base + bi[k];
}

// ----------------------------------------------- PointNetConv via WMMA -----
// Edge feature row (K=96): [h_src normalized (64) | relx | rely | 1 | 0-pad].
// Weight (as B-matrix, transposed in LDS): rows 0..63 = W_x, 64/65 = W_rel,
// 66 = bias, rest 0.  D accumulator == full per-edge message -> ds max.
template<bool HASPREV>
__global__ __launch_bounds__(128)
void conv_kernel(const float* __restrict__ pos,
                 const int*   __restrict__ nbr,
                 const float* __restrict__ hprev,   // N x 64 (pre-norm), may be unused
                 const float* __restrict__ stats,   // mean[0..63], inv at [64]
                 const float* __restrict__ W, int fanin,
                 const float* __restrict__ bias,
                 float* __restrict__ hout,
                 float* __restrict__ pc, float* __restrict__ pss)
{
    __shared__ __align__(16) _Float16 Alds[EDGES*KDIM];   // 33792 B
    __shared__ __align__(16) _Float16 Wt[HDIM*KDIM];      // 12288 B
    __shared__ float outl[DPB*HDIM];                      // per-dst running max
    __shared__ int   srcl[EDGES];
    __shared__ float rxl[EDGES], ryl[EDGES];
    __shared__ float smean[65];
    __shared__ float red1[128], red2[128];

    const int tid  = threadIdx.x;
    const int dst0 = blockIdx.x * DPB;

    // ---- edge metadata + stats staging ---------------------------------
    if (HASPREV && tid < 65) smean[tid] = stats[tid];
    for (int e = tid; e < EDGES; e += 128) {
        const int dl = e / EPD, k = e - dl*EPD;
        const int dst = dst0 + dl;
        const int src = (k < KNN) ? nbr[dst*KNN + k] : dst;   // self loop last
        srcl[e] = src;
        rxl[e]  = pos[2*src]   - pos[2*dst];
        ryl[e]  = pos[2*src+1] - pos[2*dst+1];
    }
    // ---- stage W^T (B-matrix: [n][k], contiguous K per output column) --
    if (HASPREV) {
        for (int i = tid; i < HDIM*KDIM; i += 128) {
            const int n = i / KDIM, k = i - n*KDIM;
            float v = 0.f;
            if (k < HDIM)            v = W[k*HDIM + n];
            else if (k == HDIM)      v = W[(fanin-2)*HDIM + n];
            else if (k == HDIM+1)    v = W[(fanin-1)*HDIM + n];
            else if (k == HDIM+2)    v = bias[n];
            Wt[i] = (_Float16)v;
        }
    } else {
        // only k-step 2 (cols 64..95) is ever read
        for (int i = tid; i < HDIM*32; i += 128) {
            const int n = i >> 5, k = HDIM + (i & 31);
            float v = 0.f;
            if (k == HDIM)           v = W[(fanin-2)*HDIM + n];
            else if (k == HDIM+1)    v = W[(fanin-1)*HDIM + n];
            else if (k == HDIM+2)    v = bias[n];
            Wt[n*KDIM + k] = (_Float16)v;
        }
    }
    for (int i = tid; i < DPB*HDIM; i += 128) outl[i] = -3.0e38f;
    __syncthreads();

    // ---- stage A: 4-wide quads, b128 gather + fused PairNorm -----------
    if (HASPREV) {
        const float inv = smean[64];
        for (int i = tid; i < EDGES*24; i += 128) {
            const int e = i / 24, q = i - e*24;   // quad of 4 columns
            v4h hv = {};
            if (q < 16) {
                const float4 f = *(const float4*)&hprev[srcl[e]*HDIM + q*4];
                hv[0] = (_Float16)((f.x - smean[q*4+0]) * inv);
                hv[1] = (_Float16)((f.y - smean[q*4+1]) * inv);
                hv[2] = (_Float16)((f.z - smean[q*4+2]) * inv);
                hv[3] = (_Float16)((f.w - smean[q*4+3]) * inv);
            } else if (q == 16) {
                hv[0] = (_Float16)rxl[e];
                hv[1] = (_Float16)ryl[e];
                hv[2] = (_Float16)1.f;
            }
            *(v4h*)&Alds[e*KDIM + q*4] = hv;
        }
    } else {
        // only quads 16..23 (cols 64..95) are ever read
        for (int i = tid; i < EDGES*8; i += 128) {
            const int e = i >> 3, q = 16 + (i & 7);
            v4h hv = {};
            if (q == 16) {
                hv[0] = (_Float16)rxl[e];
                hv[1] = (_Float16)ryl[e];
                hv[2] = (_Float16)1.f;
            }
            *(v4h*)&Alds[e*KDIM + q*4] = hv;
        }
    }
    __syncthreads();

    // ---- WMMA: 11 edge-tiles x 4 N-tiles, chained K-steps --------------
    const int wave = tid >> 5, lane = tid & 31;
    const int col  = lane & 15, kh = lane >> 4;
    constexpr int KS_BEGIN = HASPREV ? 0 : 2;   // layer1 features live in k-step 2
    for (int job = wave; job < ETILES*4; job += 4) {
        const int tile = job >> 2, nt = job & 3;
        v8f c = {};
#pragma unroll
        for (int s = KS_BEGIN; s < 3; ++s) {
            // A frag: lane half kh holds K = {0..7,16..23} / {8..15,24..31}
            const _Float16* ab = &Alds[(tile*16 + col)*KDIM + s*32 + kh*8];
            const v8h a0 = *(const v8h*)ab;
            const v8h a1 = *(const v8h*)(ab + 16);
            v16h a;
#pragma unroll
            for (int q = 0; q < 8; ++q) { a[q] = a0[q]; a[q+8] = a1[q]; }
            // B frag: lane half kh holds K = {0..7,16..23} / {8..15,24..31} at column col
            const _Float16* bb = &Wt[(nt*16 + col)*KDIM + s*32 + kh*8];
            const v8h b0 = *(const v8h*)bb;
            const v8h b1 = *(const v8h*)(bb + 16);
            v16h bf;
#pragma unroll
            for (int q = 0; q < 8; ++q) { bf[q] = b0[q]; bf[q+8] = b1[q]; }
            c = __builtin_amdgcn_wmma_f32_16x16x32_f16(false, a, false, bf,
                                                       (short)0, c, false, false);
        }
        // D layout: VGPR r -> edge row r + 8*kh, column col. Segment-max in LDS.
        // 8 consecutive rows span at most 2 dst segments: 1 divide, then select.
        const int rbase = tile*16 + 8*kh;
        const int dl0   = rbase / EPD;
        const int bnd   = (dl0 + 1)*EPD - rbase;      // r < bnd -> dl0 else dl0+1
        const unsigned a0addr =
            (unsigned)(unsigned long long)(&outl[dl0*HDIM + nt*16 + col]);
        const unsigned a1addr = a0addr + HDIM*4;
#pragma unroll
        for (int r = 0; r < 8; ++r) {
            const unsigned addr = (r < bnd) ? a0addr : a1addr;
            const float val = c[r];
            asm volatile("ds_max_num_f32 %0, %1" :: "v"(addr), "v"(val) : "memory");
        }
    }
    asm volatile("s_wait_dscnt 0x0" ::: "memory");
    __syncthreads();

    // ---- ReLU, store, deterministic PairNorm partials ------------------
    const int mycol = tid & 63;          // 128 % 64 == 0 -> column fixed per thread
    float cs = 0.f, ss = 0.f;
    for (int i = tid; i < DPB*HDIM; i += 128) {
        const float v = fmaxf(outl[i], 0.f);
        hout[(dst0 + (i >> 6))*HDIM + mycol] = v;
        cs += v; ss += v*v;
    }
    red1[tid] = cs; red2[tid] = ss;
    __syncthreads();
    if (tid < 64) pc[blockIdx.x*64 + tid] = red1[tid] + red1[tid+64];
    if (tid == 0) {
        float t = 0.f;
        for (int i = 0; i < 128; ++i) t += red2[i];
        pss[blockIdx.x] = t;
    }
}

// --------------------------------------- PairNorm stats finalization -------
__global__ __launch_bounds__(64)
void finalize_kernel(const float* __restrict__ pc, const float* __restrict__ pss,
                     float* __restrict__ stats) {
    __shared__ float msq[64], ssp[64];
    const int t = threadIdx.x;
    float s = 0.f;
    for (int b = 0; b < NBLK_CONV; ++b) s += pc[b*64 + t];
    const float mu = s / (float)NPTS;
    stats[t] = mu;
    msq[t]   = mu * mu;
    float ssl = 0.f;
    for (int b = t; b < NBLK_CONV; b += 64) ssl += pss[b];
    ssp[t] = ssl;
    __syncthreads();
    if (t == 0) {
        float m2 = 0.f, sst = 0.f;
        for (int i = 0; i < 64; ++i) { m2 += msq[i]; sst += ssp[i]; }
        const float meansq = sst / (float)NPTS - m2;   // mean ||xc||^2
        stats[64] = rsqrtf(EPS_PN + meansq);
    }
}

// --------------------------------------------------- global max pool -------
// max commutes with (x - mean)*inv (inv > 0): pool raw, normalize pooled.
__global__ __launch_bounds__(256)
void pool_kernel(const float* __restrict__ h, const float* __restrict__ stats,
                 float* __restrict__ g) {
    __shared__ float red[256];
    const int b = blockIdx.x, t = threadIdx.x;
    const int colp = t & 63, seg = t >> 6;
    float m = -3.0e38f;
    const int base = b*NPG + seg*(NPG/4);
    for (int i = 0; i < NPG/4; ++i)
        m = fmaxf(m, h[(base+i)*HDIM + colp]);
    red[t] = m;
    __syncthreads();
    if (t < 64) {
        const float mm = fmaxf(fmaxf(red[t], red[t+64]), fmaxf(red[t+128], red[t+192]));
        g[b*HDIM + t] = (mm - stats[t]) * stats[64];
    }
}

// ------------------------------------------------------------ MLP head -----
__global__ __launch_bounds__(64)
void head_kernel(const float* __restrict__ g,
                 const float* __restrict__ Wl1, const float* __restrict__ bl1,
                 const float* __restrict__ Wl2, const float* __restrict__ bl2,
                 float* __restrict__ out) {
    __shared__ float gv[64], t1[64];
    const int b = blockIdx.x, t = threadIdx.x;
    gv[t] = g[b*64 + t];
    __syncthreads();
    float acc = bl1[t];
    for (int k = 0; k < 64; ++k) acc += gv[k] * Wl1[k*64 + t];
    t1[t] = fmaxf(acc, 0.f);
    __syncthreads();
    if (t < 2) {
        float o = bl2[t];
        for (int k = 0; k < 64; ++k) o += t1[k] * Wl2[k*2 + t];
        out[b*2 + t] = o;
    }
}

// ---------------------------------------------------------------------------
extern "C" void kernel_launch(void* const* d_in, const int* in_sizes, int n_in,
                              void* d_out, int out_size, void* d_ws, size_t ws_size,
                              hipStream_t stream) {
    (void)in_sizes; (void)n_in; (void)out_size; (void)ws_size;
    const float* pos = (const float*)d_in[0];
    // d_in[1] = batch (implicit by layout) — unused
    const float* W1  = (const float*)d_in[2];
    const float* b1  = (const float*)d_in[3];
    const float* W2  = (const float*)d_in[4];
    const float* b2  = (const float*)d_in[5];
    const float* W3  = (const float*)d_in[6];
    const float* b3  = (const float*)d_in[7];
    const float* Wl1 = (const float*)d_in[8];
    const float* bl1 = (const float*)d_in[9];
    const float* Wl2 = (const float*)d_in[10];
    const float* bl2 = (const float*)d_in[11];
    float* out = (float*)d_out;

    char* ws = (char*)d_ws;
    size_t off = 0;
    auto alloc = [&](size_t bytes) {
        void* p = ws + off; off = (off + bytes + 255) & ~(size_t)255; return p;
    };
    int*   nbr = (int*)  alloc(sizeof(int)  * NPTS * KNN);
    float* hA  = (float*)alloc(sizeof(float)* NPTS * HDIM);
    float* hB  = (float*)alloc(sizeof(float)* NPTS * HDIM);
    float* pc  = (float*)alloc(sizeof(float)* NBLK_CONV * 64);
    float* pss = (float*)alloc(sizeof(float)* NBLK_CONV);
    float* st0 = (float*)alloc(sizeof(float)* 128);
    float* st1 = (float*)alloc(sizeof(float)* 128);
    float* st2 = (float*)alloc(sizeof(float)* 128);
    float* g   = (float*)alloc(sizeof(float)* B_GR * HDIM);

    knn_kernel<<<B_GR*(NPG/128), 128, 0, stream>>>(pos, nbr);

    conv_kernel<false><<<NBLK_CONV, 128, 0, stream>>>(pos, nbr, nullptr, st0,
                                                      W1, 2,  b1, hA, pc, pss);
    finalize_kernel<<<1, 64, 0, stream>>>(pc, pss, st0);

    conv_kernel<true><<<NBLK_CONV, 128, 0, stream>>>(pos, nbr, hA, st0,
                                                     W2, 66, b2, hB, pc, pss);
    finalize_kernel<<<1, 64, 0, stream>>>(pc, pss, st1);

    conv_kernel<true><<<NBLK_CONV, 128, 0, stream>>>(pos, nbr, hB, st1,
                                                     W3, 66, b3, hA, pc, pss);
    finalize_kernel<<<1, 64, 0, stream>>>(pc, pss, st2);

    pool_kernel<<<B_GR, 256, 0, stream>>>(hA, st2, g);
    head_kernel<<<B_GR, 64, 0, stream>>>(g, Wl1, bl1, Wl2, bl2, out);
}
